// Block_3401614099134
// MI455X (gfx1250) — matrix operand order repaired and checked by reading, not compiled, not run on hardware
//
#include <hip/hip_runtime.h>
#include <math.h>

typedef __attribute__((ext_vector_type(16))) __bf16 v16bf;
typedef __attribute__((ext_vector_type(8)))  __bf16 v8bf;
typedef __attribute__((ext_vector_type(8)))  float  v8f;

// ---------------------------------------------------------------------------
// WMMA helpers (CDNA5 gfx1250, wave32)
// ---------------------------------------------------------------------------
__device__ __forceinline__ v8f wmma_bf16(v16bf a, v16bf b, v8f c) {
  // v_wmma_f32_16x16x32_bf16: D = A(16x32) * B(32x16) + C(16x16 f32)
  return __builtin_amdgcn_wmma_f32_16x16x32_bf16(
      /*neg_a=*/false, a, /*neg_b=*/false, b,
      /*c_mod=*/(short)0, c, /*reuse_a=*/false, /*reuse_b=*/false);
}

// Load a 16x32 bf16 fragment (A layout; B 32x16 column-major uses the same
// pattern when the LDS tile is stored [N][K] with K contiguous).
// ISA layout: lanes 0-15 row M=lane, K chunks [0..7],[16..23];
//             lanes 16-31 row M=lane-16, K chunks [8..15],[24..31].
__device__ __forceinline__ v16bf load_frag(const __bf16* p, int stride) {
  const int lane = threadIdx.x & 31;
  const __bf16* r = p + (lane & 15) * stride + ((lane >> 4) << 3);
  v8bf lo = *(const v8bf*)(r);
  v8bf hi = *(const v8bf*)(r + 16);
  v16bf out;
#pragma unroll
  for (int i = 0; i < 8; ++i) { out[i] = lo[i]; out[8 + i] = hi[i]; }
  return out;
}

// ---------------------------------------------------------------------------
// Async global->LDS copy (gfx1250 TDM-adjacent async path, ASYNCcnt).
// 16 bytes per lane per instruction; LDS byte offset = low 32 bits of the
// generic LDS pointer (ISA flat-address truncation rule).
// ---------------------------------------------------------------------------
__device__ __forceinline__ void async_ld16(const void* g, void* l) {
  asm volatile("global_load_async_to_lds_b128 %0, %1, off"
               :
               : "v"((unsigned)(unsigned long long)(uintptr_t)l),
                 "v"((unsigned long long)(uintptr_t)g)
               : "memory");
}
__device__ __forceinline__ void wait_async() {
  asm volatile("s_wait_asynccnt 0" ::: "memory");
}

// ---------------------------------------------------------------------------
// RMSNorm: one block per token, D=1024
// ---------------------------------------------------------------------------
__global__ void __launch_bounds__(256)
rmsnorm_kernel(const float* __restrict__ x, const float* __restrict__ w,
               __bf16* __restrict__ out, int D) {
  const int n = blockIdx.x;
  const int tid = threadIdx.x;
  float ss = 0.f;
  for (int d = tid; d < D; d += 256) {
    float v = x[(long)n * D + d];
    ss += v * v;
  }
#pragma unroll
  for (int m = 16; m >= 1; m >>= 1) ss += __shfl_xor(ss, m, 32);
  __shared__ float red[8];
  if ((tid & 31) == 0) red[tid >> 5] = ss;
  __syncthreads();
  __shared__ float srms;
  if (tid == 0) {
    float t = 0.f;
    for (int i = 0; i < 8; ++i) t += red[i];
    srms = rsqrtf(t / (float)D + 1e-6f);
  }
  __syncthreads();
  const float r = srms;
  for (int d = tid; d < D; d += 256)
    out[(long)n * D + d] = (__bf16)(x[(long)n * D + d] * r * w[d]);
}

// ---------------------------------------------------------------------------
// Generic GEMM: C[m,n] = sum_k A_bf16[m,k] * W_f32[n,k] + bias[n] (+resid)
// Tile 128x128, BK=32, 256 threads (8 waves, 2x4 wave grid).
// A tile staged via async global->LDS; W tile converted f32->bf16 on VALU.
// ---------------------------------------------------------------------------
template <bool RESID>
__global__ void __launch_bounds__(256)
gemm_bf16_kernel(const __bf16* __restrict__ A, const float* __restrict__ W,
                 const float* __restrict__ bias, const float* __restrict__ resid,
                 float* __restrict__ C, int M, int N, int K) {
  __shared__ __align__(16) __bf16 As[128 * 48];
  __shared__ __align__(16) __bf16 Ws[128 * 48];
  const int tid = threadIdx.x, wave = tid >> 5, lane = tid & 31;
  const int m0 = blockIdx.y * 128, n0 = blockIdx.x * 128;
  const int wm = (wave >> 2) * 64, wn = (wave & 3) * 32;
  v8f zero = {};
  v8f acc[4][2];
#pragma unroll
  for (int i = 0; i < 4; ++i) { acc[i][0] = zero; acc[i][1] = zero; }

  for (int k0 = 0; k0 < K; k0 += 32) {
    {  // A tile: 128x32 bf16, async copy (no VGPR staging)
      int r = tid >> 2, c = (tid & 3) * 8;
      async_ld16(&A[(long)(m0 + r) * K + k0 + c], &As[r * 48 + c]);
      async_ld16(&A[(long)(m0 + r + 64) * K + k0 + c], &As[(r + 64) * 48 + c]);
    }
    {  // W tile: 128x32 f32 -> bf16 (VALU convert path)
      int r = tid >> 3, c = (tid & 7) * 4;
#pragma unroll
      for (int p = 0; p < 4; ++p) {
        float4 wv = *(const float4*)&W[(long)(n0 + r + p * 32) * K + k0 + c];
        __bf16* d = &Ws[(r + p * 32) * 48 + c];
        d[0] = (__bf16)wv.x; d[1] = (__bf16)wv.y;
        d[2] = (__bf16)wv.z; d[3] = (__bf16)wv.w;
      }
      if (k0 + 32 < K)  // prefetch next K-slab of W into GL2
        __builtin_prefetch(&W[(long)(n0 + r) * K + k0 + 32 + c], 0, 1);
    }
    wait_async();
    __syncthreads();
    v16bf bf0 = load_frag(&Ws[(wn) * 48], 48);
    v16bf bf1 = load_frag(&Ws[(wn + 16) * 48], 48);
    v16bf af[4];
#pragma unroll
    for (int i = 0; i < 4; ++i) af[i] = load_frag(&As[(wm + i * 16) * 48], 48);
#pragma unroll
    for (int i = 0; i < 4; ++i) {
      acc[i][0] = wmma_bf16(af[i], bf0, acc[i][0]);
      acc[i][1] = wmma_bf16(af[i], bf1, acc[i][1]);
    }
    __syncthreads();
  }
  const int crow = (lane >> 4) << 3, ccol = lane & 15;
#pragma unroll
  for (int i = 0; i < 4; ++i)
#pragma unroll
    for (int j = 0; j < 2; ++j) {
      int gn = n0 + wn + j * 16 + ccol;
      float b = bias[gn];
#pragma unroll
      for (int jj = 0; jj < 8; ++jj) {
        int gm = m0 + wm + i * 16 + crow + jj;
        float v = acc[i][j][jj] + b;
        if (RESID) v += resid[(long)gm * N + gn];
        C[(long)gm * N + gn] = v;
      }
    }
}

// ---------------------------------------------------------------------------
// RoPE: qkv f32 [N][H][3][64] -> Q/K/V bf16 [B*H][T][64]
// ---------------------------------------------------------------------------
__global__ void __launch_bounds__(256)
rope_kernel(const float* __restrict__ qkv, __bf16* __restrict__ Q,
            __bf16* __restrict__ K, __bf16* __restrict__ V, int T, int Hh) {
  int idx = blockIdx.x * 256 + threadIdx.x;          // over N*H*32
  int i = idx & 31;
  int h = (idx >> 5) & (Hh - 1);
  int n = idx >> 9;                                   // / (Hh*32) with Hh=16
  int t = n & (T - 1);
  long base = (long)n * (Hh * 192) + h * 192;
  float q1 = qkv[base + i],       q2 = qkv[base + 32 + i];
  float k1 = qkv[base + 64 + i],  k2 = qkv[base + 96 + i];
  float v1 = qkv[base + 128 + i], v2 = qkv[base + 160 + i];
  float theta = powf(10000.f, -(float)i / 32.f);
  float f = (float)t * theta;
  float c = __cosf(f), s = __sinf(f);
  long o = ((long)((n / T) * Hh + h) * T + t) * 64;
  Q[o + i]      = (__bf16)(q1 * c - q2 * s);
  Q[o + 32 + i] = (__bf16)(q1 * s + q2 * c);
  K[o + i]      = (__bf16)(k1 * c - k2 * s);
  K[o + 32 + i] = (__bf16)(k1 * s + k2 * c);
  V[o + i]      = (__bf16)v1;
  V[o + 32 + i] = (__bf16)v2;
}

// ---------------------------------------------------------------------------
// Flash-style causal attention. Q/K/V bf16 [B*H][T][64]; O bf16 [B][T][D].
// 128 q-rows per block, stream 32-wide kv blocks. Wave w owns rows w*16..+15.
// ---------------------------------------------------------------------------
__global__ void __launch_bounds__(256)
attn_kernel(const __bf16* __restrict__ Q, const __bf16* __restrict__ Km,
            const __bf16* __restrict__ V, __bf16* __restrict__ O,
            int T, int Hh, float scale) {
  const int bh = blockIdx.y;
  const int q0 = blockIdx.x * 128;
  const int tid = threadIdx.x, wave = tid >> 5, lane = tid & 31;
  __shared__ __align__(16) __bf16 Qs[128 * 72];
  __shared__ __align__(16) __bf16 Ks[32 * 72];
  __shared__ __align__(16) __bf16 Vs[32 * 72];
  __shared__ __align__(16) __bf16 Ps[128 * 48];

  for (int idx = tid; idx < 128 * 8; idx += 256) {  // Q tile, async
    int r = idx >> 3, c = (idx & 7) * 8;
    async_ld16(&Q[((long)bh * T + q0 + r) * 64 + c], &Qs[r * 72 + c]);
  }
  const int wm = wave * 16;
  const int crow = (lane >> 4) << 3, ccol = lane & 15;
  v8f zero = {};
  v8f o[4];
#pragma unroll
  for (int j = 0; j < 4; ++j) o[j] = zero;
  float m_run[8], l_run[8];
#pragma unroll
  for (int jj = 0; jj < 8; ++jj) { m_run[jj] = -3.0e38f; l_run[jj] = 0.f; }

  const int nkb = (q0 + 128) >> 5;  // causal: only kv blocks with base <= q0+96
  for (int kb = 0; kb < nkb; ++kb) {
    {  // K/V block 32x64, async
      int r = tid >> 3, c = (tid & 7) * 8;
      async_ld16(&Km[((long)bh * T + kb * 32 + r) * 64 + c], &Ks[r * 72 + c]);
      async_ld16(&V[((long)bh * T + kb * 32 + r) * 64 + c], &Vs[r * 72 + c]);
    }
    wait_async();
    __syncthreads();
    v8f s0 = zero, s1 = zero;
#pragma unroll
    for (int kk = 0; kk < 2; ++kk) {  // hd=64 -> two K-chunks of 32
      v16bf a  = load_frag(&Qs[wm * 72 + kk * 32], 72);
      v16bf b0 = load_frag(&Ks[0 * 72 + kk * 32], 72);
      v16bf b1 = load_frag(&Ks[16 * 72 + kk * 32], 72);
      s0 = wmma_bf16(a, b0, s0);
      s1 = wmma_bf16(a, b1, s1);
    }
    // online softmax; C-fragment row = crow+jj, col = ccol (per half-wave)
#pragma unroll
    for (int jj = 0; jj < 8; ++jj) {
      int rowg = q0 + wm + crow + jj;
      int c0 = kb * 32 + ccol, c1 = c0 + 16;
      float v0 = s0[jj] * scale, v1 = s1[jj] * scale;
      if (c0 > rowg) v0 = -1.0e9f;
      if (c1 > rowg) v1 = -1.0e9f;
      float mx = fmaxf(v0, v1);
#pragma unroll
      for (int d = 8; d >= 1; d >>= 1) mx = fmaxf(mx, __shfl_xor(mx, d, 32));
      float mn = fmaxf(m_run[jj], mx);
      float alpha = __expf(m_run[jj] - mn);
      float p0 = __expf(v0 - mn), p1 = __expf(v1 - mn);
      float rs = p0 + p1;
#pragma unroll
      for (int d = 8; d >= 1; d >>= 1) rs += __shfl_xor(rs, d, 32);
      l_run[jj] = l_run[jj] * alpha + rs;
      m_run[jj] = mn;
      o[0][jj] *= alpha; o[1][jj] *= alpha; o[2][jj] *= alpha; o[3][jj] *= alpha;
      Ps[(wm + crow + jj) * 48 + ccol]      = (__bf16)p0;
      Ps[(wm + crow + jj) * 48 + 16 + ccol] = (__bf16)p1;
    }
    __syncthreads();
    v16bf pa = load_frag(&Ps[wm * 48], 48);
    v16bf vb[4];
#pragma unroll
    for (int j = 0; j < 4; ++j)
#pragma unroll
      for (int s = 0; s < 16; ++s)
        vb[j][s] = Vs[(((lane >> 4) << 3) + (s & 7) + ((s >> 3) << 4)) * 72 + j * 16 + (lane & 15)];
#pragma unroll
    for (int j = 0; j < 4; ++j) o[j] = wmma_bf16(pa, vb[j], o[j]);
    __syncthreads();
  }
  const int b = bh / Hh, h = bh % Hh;
#pragma unroll
  for (int j = 0; j < 4; ++j)
#pragma unroll
    for (int jj = 0; jj < 8; ++jj) {
      int tok = q0 + wm + crow + jj;
      float ov = o[j][jj] / l_run[jj];
      O[((long)(b * T + tok)) * 1024 + h * 64 + j * 16 + ccol] = (__bf16)ov;
    }
}

// ---------------------------------------------------------------------------
// Router: per-token logits over 8 experts, softmax, top-2, counts
// ---------------------------------------------------------------------------
__global__ void __launch_bounds__(256)
router_kernel(const __bf16* __restrict__ xn, const float* __restrict__ gw,
              float* __restrict__ probs, int* __restrict__ tidx,
              float* __restrict__ tgate, int* __restrict__ counts, int D) {
  const int n = blockIdx.x, tid = threadIdx.x;
  float part[8] = {0, 0, 0, 0, 0, 0, 0, 0};
  for (int d = tid; d < D; d += 256) {
    float xv = (float)xn[(long)n * D + d];
#pragma unroll
    for (int e = 0; e < 8; ++e) part[e] += xv * gw[e * D + d];
  }
#pragma unroll
  for (int e = 0; e < 8; ++e)
#pragma unroll
    for (int m = 16; m >= 1; m >>= 1) part[e] += __shfl_xor(part[e], m, 32);
  __shared__ float red[8][8];
  if ((tid & 31) == 0)
    for (int e = 0; e < 8; ++e) red[tid >> 5][e] = part[e];
  __syncthreads();
  if (tid == 0) {
    float lg[8];
    for (int e = 0; e < 8; ++e) {
      float s = 0.f;
      for (int w = 0; w < 8; ++w) s += red[w][e];
      lg[e] = s;
    }
    float mx = lg[0];
    for (int e = 1; e < 8; ++e) mx = fmaxf(mx, lg[e]);
    float sum = 0.f;
    for (int e = 0; e < 8; ++e) { lg[e] = __expf(lg[e] - mx); sum += lg[e]; }
    float inv = 1.f / sum;
    int i0 = 0; float v0 = -1.f;
    for (int e = 0; e < 8; ++e) {
      float pv = lg[e] * inv;
      probs[n * 8 + e] = pv;
      if (pv > v0) { v0 = pv; i0 = e; }
    }
    int i1 = 0; float v1 = -1.f;
    for (int e = 0; e < 8; ++e) {
      if (e == i0) continue;
      float pv = lg[e] * inv;
      if (pv > v1) { v1 = pv; i1 = e; }
    }
    tidx[n * 2] = i0; tidx[n * 2 + 1] = i1;
    tgate[n * 2] = v0; tgate[n * 2 + 1] = v1;
    atomicAdd(&counts[i0], 1);
    atomicAdd(&counts[i1], 1);
  }
}

__global__ void zero_counts_kernel(int* counts) {
  if (threadIdx.x < 8) counts[threadIdx.x] = 0;
}

__global__ void scan_kernel(const int* counts, int* off, int* nxt) {
  if (threadIdx.x == 0) {
    int a = 0;
    for (int e = 0; e < 8; ++e) { off[e] = a; nxt[e] = a; a += counts[e]; }
  }
}

__global__ void scatter_kernel(const int* __restrict__ tidx, int* nxt,
                               int* __restrict__ rows, int* __restrict__ slots,
                               int N) {
  int n = blockIdx.x * 256 + threadIdx.x;
  if (n >= N) return;
#pragma unroll
  for (int k = 0; k < 2; ++k) {
    int e = tidx[n * 2 + k];
    int s = atomicAdd(&nxt[e], 1);
    rows[s] = n;
    slots[n * 2 + k] = s;
  }
}

// ---------------------------------------------------------------------------
// MoE GEMM-1: compacted rows; dual B (w1,w3); fused SiLU*gate -> bf16 H
// Out-of-range rows clamp to a valid row (async copy can't zero-fill);
// their outputs are discarded by the lm<count epilogue guard.
// ---------------------------------------------------------------------------
__global__ void __launch_bounds__(256)
moe_ffn1_kernel(const __bf16* __restrict__ X, const float* __restrict__ ew1,
                const float* __restrict__ ew3, const float* __restrict__ eb1,
                const float* __restrict__ eb3, const int* __restrict__ rows,
                const int* __restrict__ cnt, const int* __restrict__ off,
                __bf16* __restrict__ Hbuf, int D, int I, int NA) {
  const int e = blockIdx.z;
  const int count = cnt[e];
  const int m0 = blockIdx.y * 128;
  if (m0 >= count) return;
  const int n0 = blockIdx.x * 128;
  const int base = off[e];
  __shared__ __align__(16) __bf16 As[128 * 48];
  __shared__ __align__(16) __bf16 W1s[128 * 48];
  __shared__ __align__(16) __bf16 W3s[128 * 48];
  const float* W1 = ew1 + (long)e * I * D;
  const float* W3 = ew3 + (long)e * I * D;
  const int tid = threadIdx.x, wave = tid >> 5, lane = tid & 31;
  const int wm = (wave >> 2) * 64, wn = (wave & 3) * 32;
  // gather indices (clamped) are loop-invariant
  const int r_a = tid >> 2, c_a = (tid & 3) * 8;
  int tok0 = rows[min(base + m0 + r_a, NA - 1)];
  int tok1 = rows[min(base + m0 + r_a + 64, NA - 1)];
  v8f zero = {};
  v8f a1[4][2], a3[4][2];
#pragma unroll
  for (int i = 0; i < 4; ++i) { a1[i][0] = a1[i][1] = zero; a3[i][0] = a3[i][1] = zero; }

  for (int k0 = 0; k0 < D; k0 += 32) {
    async_ld16(&X[(long)tok0 * D + k0 + c_a], &As[r_a * 48 + c_a]);
    async_ld16(&X[(long)tok1 * D + k0 + c_a], &As[(r_a + 64) * 48 + c_a]);
    {
      int r = tid >> 3, c = (tid & 7) * 4;
#pragma unroll
      for (int p = 0; p < 4; ++p) {
        float4 w1v = *(const float4*)&W1[(long)(n0 + r + p * 32) * D + k0 + c];
        float4 w3v = *(const float4*)&W3[(long)(n0 + r + p * 32) * D + k0 + c];
        __bf16* d1 = &W1s[(r + p * 32) * 48 + c];
        __bf16* d3 = &W3s[(r + p * 32) * 48 + c];
        d1[0] = (__bf16)w1v.x; d1[1] = (__bf16)w1v.y; d1[2] = (__bf16)w1v.z; d1[3] = (__bf16)w1v.w;
        d3[0] = (__bf16)w3v.x; d3[1] = (__bf16)w3v.y; d3[2] = (__bf16)w3v.z; d3[3] = (__bf16)w3v.w;
      }
      if (k0 + 32 < D) {
        __builtin_prefetch(&W1[(long)(n0 + r) * D + k0 + 32 + c], 0, 1);
        __builtin_prefetch(&W3[(long)(n0 + r) * D + k0 + 32 + c], 0, 1);
      }
    }
    wait_async();
    __syncthreads();
    v16bf b10 = load_frag(&W1s[(wn) * 48], 48);
    v16bf b11 = load_frag(&W1s[(wn + 16) * 48], 48);
    v16bf b30 = load_frag(&W3s[(wn) * 48], 48);
    v16bf b31 = load_frag(&W3s[(wn + 16) * 48], 48);
    v16bf af[4];
#pragma unroll
    for (int i = 0; i < 4; ++i) af[i] = load_frag(&As[(wm + i * 16) * 48], 48);
#pragma unroll
    for (int i = 0; i < 4; ++i) {
      a1[i][0] = wmma_bf16(af[i], b10, a1[i][0]);
      a1[i][1] = wmma_bf16(af[i], b11, a1[i][1]);
      a3[i][0] = wmma_bf16(af[i], b30, a3[i][0]);
      a3[i][1] = wmma_bf16(af[i], b31, a3[i][1]);
    }
    __syncthreads();
  }
  const int crow = (lane >> 4) << 3, ccol = lane & 15;
#pragma unroll
  for (int i = 0; i < 4; ++i)
#pragma unroll
    for (int j = 0; j < 2; ++j) {
      int gn = n0 + wn + j * 16 + ccol;
      float b1 = eb1[e * I + gn], b3 = eb3[e * I + gn];
#pragma unroll
      for (int jj = 0; jj < 8; ++jj) {
        int lm = m0 + wm + i * 16 + crow + jj;
        if (lm < count) {
          float av = a1[i][j][jj] + b1;
          float uv = a3[i][j][jj] + b3;
          float hv = (av / (1.f + __expf(-av))) * uv;  // silu(a) * u
          Hbuf[(long)(base + lm) * I + gn] = (__bf16)hv;
        }
      }
    }
}

// ---------------------------------------------------------------------------
// MoE GEMM-2: y[slot] = H[slot] @ w2[e]^T + eb2[e]
// ---------------------------------------------------------------------------
__global__ void __launch_bounds__(256)
moe_ffn2_kernel(const __bf16* __restrict__ Hbuf, const float* __restrict__ ew2,
                const float* __restrict__ eb2, const int* __restrict__ cnt,
                const int* __restrict__ off, float* __restrict__ Y,
                int D, int I, int NA) {
  const int e = blockIdx.z;
  const int count = cnt[e];
  const int m0 = blockIdx.y * 128;
  if (m0 >= count) return;
  const int n0 = blockIdx.x * 128;
  const int base = off[e];
  __shared__ __align__(16) __bf16 As[128 * 48];
  __shared__ __align__(16) __bf16 Ws[128 * 48];
  const float* W2 = ew2 + (long)e * D * I;
  const int tid = threadIdx.x, wave = tid >> 5, lane = tid & 31;
  const int wm = (wave >> 2) * 64, wn = (wave & 3) * 32;
  const int r_a = tid >> 2, c_a = (tid & 3) * 8;
  const long row0 = min(base + m0 + r_a, NA - 1);
  const long row1 = min(base + m0 + r_a + 64, NA - 1);
  v8f zero = {};
  v8f acc[4][2];
#pragma unroll
  for (int i = 0; i < 4; ++i) { acc[i][0] = zero; acc[i][1] = zero; }

  for (int k0 = 0; k0 < I; k0 += 32) {
    async_ld16(&Hbuf[row0 * I + k0 + c_a], &As[r_a * 48 + c_a]);
    async_ld16(&Hbuf[row1 * I + k0 + c_a], &As[(r_a + 64) * 48 + c_a]);
    {
      int r = tid >> 3, c = (tid & 7) * 4;
#pragma unroll
      for (int p = 0; p < 4; ++p) {
        float4 wv = *(const float4*)&W2[(long)(n0 + r + p * 32) * I + k0 + c];
        __bf16* d = &Ws[(r + p * 32) * 48 + c];
        d[0] = (__bf16)wv.x; d[1] = (__bf16)wv.y; d[2] = (__bf16)wv.z; d[3] = (__bf16)wv.w;
      }
      if (k0 + 32 < I)
        __builtin_prefetch(&W2[(long)(n0 + r) * I + k0 + 32 + c], 0, 1);
    }
    wait_async();
    __syncthreads();
    v16bf b0 = load_frag(&Ws[(wn) * 48], 48);
    v16bf b1 = load_frag(&Ws[(wn + 16) * 48], 48);
    v16bf af[4];
#pragma unroll
    for (int i = 0; i < 4; ++i) af[i] = load_frag(&As[(wm + i * 16) * 48], 48);
#pragma unroll
    for (int i = 0; i < 4; ++i) {
      acc[i][0] = wmma_bf16(af[i], b0, acc[i][0]);
      acc[i][1] = wmma_bf16(af[i], b1, acc[i][1]);
    }
    __syncthreads();
  }
  const int crow = (lane >> 4) << 3, ccol = lane & 15;
#pragma unroll
  for (int i = 0; i < 4; ++i)
#pragma unroll
    for (int j = 0; j < 2; ++j) {
      int gn = n0 + wn + j * 16 + ccol;
      float b = eb2[e * D + gn];
#pragma unroll
      for (int jj = 0; jj < 8; ++jj) {
        int lm = m0 + wm + i * 16 + crow + jj;
        if (lm < count) Y[(long)(base + lm) * D + gn] = acc[i][j][jj] + b;
      }
    }
}

__global__ void __launch_bounds__(256)
combine_kernel(const float* __restrict__ h1, const float* __restrict__ Y,
               const float* __restrict__ gates, const int* __restrict__ slots,
               float* __restrict__ out, int ND, int D) {
  int i = blockIdx.x * 256 + threadIdx.x;
  if (i >= ND) return;
  int n = i / D, d = i - n * D;
  float v = h1[i];
  v += gates[n * 2]     * Y[(long)slots[n * 2] * D + d];
  v += gates[n * 2 + 1] * Y[(long)slots[n * 2 + 1] * D + d];
  out[i] = v;
}

__global__ void __launch_bounds__(256)
aux_kernel(const float* __restrict__ probs, const int* __restrict__ counts,
           float* __restrict__ out_aux, int N) {
  const int tid = threadIdx.x;
  float part[8] = {0, 0, 0, 0, 0, 0, 0, 0};
  for (int n = tid; n < N; n += 256)
#pragma unroll
    for (int e = 0; e < 8; ++e) part[e] += probs[n * 8 + e];
#pragma unroll
  for (int e = 0; e < 8; ++e)
#pragma unroll
    for (int m = 16; m >= 1; m >>= 1) part[e] += __shfl_xor(part[e], m, 32);
  __shared__ float red[8][8];
  if ((tid & 31) == 0)
    for (int e = 0; e < 8; ++e) red[tid >> 5][e] = part[e];
  __syncthreads();
  if (tid == 0) {
    float aux = 0.f;
    for (int e = 0; e < 8; ++e) {
      float P = 0.f;
      for (int w = 0; w < 8; ++w) P += red[w][e];
      P /= (float)N;
      float f = (float)counts[e] / (float)(N * 2);
      aux += f * P;
    }
    out_aux[0] = 0.01f * 8.f * aux;
  }
}

// ---------------------------------------------------------------------------
extern "C" void kernel_launch(void* const* d_in, const int* in_sizes, int n_in,
                              void* d_out, int out_size, void* d_ws, size_t ws_size,
                              hipStream_t stream) {
  const int B = 2, T = 1024, D = 1024, Hh = 16, E = 8, I = 4096;
  const int N = B * T;
  const float* x      = (const float*)d_in[0];
  const float* qkv_w  = (const float*)d_in[1];
  const float* qkv_b  = (const float*)d_in[2];
  const float* out_w  = (const float*)d_in[3];
  const float* out_b  = (const float*)d_in[4];
  const float* ln1_w  = (const float*)d_in[5];
  const float* ln2_w  = (const float*)d_in[6];
  const float* gate_w = (const float*)d_in[7];
  const float* ew1    = (const float*)d_in[8];
  const float* eb1    = (const float*)d_in[9];
  const float* ew2    = (const float*)d_in[10];
  const float* eb2    = (const float*)d_in[11];
  const float* ew3    = (const float*)d_in[12];
  const float* eb3    = (const float*)d_in[13];
  float* out = (float*)d_out;

  char* wp = (char*)d_ws;
  auto alloc = [&](size_t bytes) -> void* {
    void* p = (void*)wp;
    wp += (bytes + 255) & ~(size_t)255;
    return p;
  };
  __bf16* xn1   = (__bf16*)alloc((size_t)N * D * 2);
  float*  qkv   = (float*) alloc((size_t)N * 3 * D * 4);
  __bf16* Qb    = (__bf16*)alloc((size_t)N * D * 2);
  __bf16* Kb    = (__bf16*)alloc((size_t)N * D * 2);
  __bf16* Vb    = (__bf16*)alloc((size_t)N * D * 2);
  __bf16* Ob    = (__bf16*)alloc((size_t)N * D * 2);
  float*  h1    = (float*) alloc((size_t)N * D * 4);
  __bf16* xn2   = (__bf16*)alloc((size_t)N * D * 2);
  float*  probs = (float*) alloc((size_t)N * E * 4);
  int*    tidx  = (int*)   alloc((size_t)N * 2 * 4);
  float*  tgate = (float*) alloc((size_t)N * 2 * 4);
  int*    counts= (int*)   alloc(256);
  int*    offs  = (int*)   alloc(256);
  int*    nxt   = (int*)   alloc(256);
  int*    rows  = (int*)   alloc((size_t)N * 2 * 4);
  int*    slots = (int*)   alloc((size_t)N * 2 * 4);
  __bf16* Hbuf  = (__bf16*)alloc((size_t)N * 2 * I * 2);
  float*  Y     = (float*) alloc((size_t)N * 2 * D * 4);

  // ---- attention path ----
  rmsnorm_kernel<<<N, 256, 0, stream>>>(x, ln1_w, xn1, D);
  gemm_bf16_kernel<false><<<dim3(3 * D / 128, N / 128), 256, 0, stream>>>(
      xn1, qkv_w, qkv_b, nullptr, qkv, N, 3 * D, D);
  rope_kernel<<<(N * Hh * 32) / 256, 256, 0, stream>>>(qkv, Qb, Kb, Vb, T, Hh);
  attn_kernel<<<dim3(T / 128, B * Hh), 256, 0, stream>>>(
      Qb, Kb, Vb, Ob, T, Hh, 1.0f / 32.0f);  // scale = D^-0.5
  gemm_bf16_kernel<true><<<dim3(D / 128, N / 128), 256, 0, stream>>>(
      Ob, out_w, out_b, x, h1, N, D, D);     // fused +x residual -> h1

  // ---- MoE path ----
  rmsnorm_kernel<<<N, 256, 0, stream>>>(h1, ln2_w, xn2, D);
  zero_counts_kernel<<<1, 32, 0, stream>>>(counts);
  router_kernel<<<N, 256, 0, stream>>>(xn2, gate_w, probs, tidx, tgate, counts, D);
  scan_kernel<<<1, 32, 0, stream>>>(counts, offs, nxt);
  scatter_kernel<<<(N + 255) / 256, 256, 0, stream>>>(tidx, nxt, rows, slots, N);
  moe_ffn1_kernel<<<dim3(I / 128, N / 128, E), 256, 0, stream>>>(
      xn2, ew1, ew3, eb1, eb3, rows, counts, offs, Hbuf, D, I, N * 2);
  moe_ffn2_kernel<<<dim3(D / 128, N / 128, E), 256, 0, stream>>>(
      Hbuf, ew2, eb2, counts, offs, Y, D, I, N * 2);
  combine_kernel<<<(N * D) / 256, 256, 0, stream>>>(h1, Y, tgate, slots, out, N * D, D);
  aux_kernel<<<1, 256, 0, stream>>>(probs, counts, out + (size_t)N * D, N);
}
